// GCNembedder_5059471475038
// MI455X (gfx1250) — matrix operand order repaired and checked by reading, not compiled, not run on hardware
//
#include <hip/hip_runtime.h>
#include <hip/hip_bf16.h>

typedef __attribute__((ext_vector_type(16))) _Float16 v16h;
typedef __attribute__((ext_vector_type(8)))  _Float16 v8h;
typedef __attribute__((ext_vector_type(8)))  float    v8f;

#define KNN   20
#define NP    4096          // points per batch
#define NBAT  4
#define BNODE 16384         // NBAT*NP
#define ME    (BNODE*KNN)   // 327680 edge rows
#define EPS_BN 1e-5f

#define WMMA_F16(a,b,c) __builtin_amdgcn_wmma_f32_16x16x32_f16(false,(a),false,(b),(short)0,(c),false,false)

// ---------------------------------------------------------------------------
// x0 = cat([feat, pos]) : swap halves of the 6 input channels
// ---------------------------------------------------------------------------
__global__ void build_x0(const float* __restrict__ P, float* __restrict__ X0) {
  int t = blockIdx.x * 256 + threadIdx.x;
  if (t >= BNODE * 6) return;
  int r = t / 6, c = t - r * 6;
  X0[t] = (c < 3) ? P[(size_t)r * 6 + 3 + c] : P[(size_t)r * 6 + c - 3];
}

// ---------------------------------------------------------------------------
// kNN (k=20, self included): one thread per query, candidate tiles in LDS,
// register-resident insertion sort (all static indices -> stays in VGPRs).
// ---------------------------------------------------------------------------
template <int D>
__global__ __launch_bounds__(128) void knn_kernel(const float* __restrict__ X,
                                                  int* __restrict__ IDX) {
  __shared__ float tile[128 * D];
  const int tid = threadIdx.x;
  const int q   = blockIdx.x * 128 + tid;       // global query row
  const int b   = q >> 12;                      // /NP
  const int n   = q & (NP - 1);
  const float* xb = X + (size_t)b * NP * D;

  float qv[D];
#pragma unroll
  for (int c = 0; c < D; ++c) qv[c] = xb[(size_t)n * D + c];

  float kd[KNN];
  int   ki[KNN];
#pragma unroll
  for (int s = 0; s < KNN; ++s) { kd[s] = 3.4e38f; ki[s] = 0; }

  for (int t0 = 0; t0 < NP; t0 += 128) {
    __syncthreads();
    for (int e = tid; e < 128 * D; e += 128) tile[e] = xb[(size_t)t0 * D + e];
    __syncthreads();
    for (int j = 0; j < 128; ++j) {
      float d = 0.f;
#pragma unroll
      for (int c = 0; c < D; ++c) {
        float df = qv[c] - tile[j * D + c];
        d = fmaf(df, df, d);
      }
      if (d < kd[KNN - 1]) {
        int cand = t0 + j;
        int pos  = KNN - 1;
#pragma unroll
        for (int s = KNN - 2; s >= 0; --s)
          if (d < kd[s]) pos = s;               // insert after equals (stable)
#pragma unroll
        for (int s = KNN - 1; s > 0; --s) {
          bool mv = (s > pos);
          kd[s] = mv ? kd[s - 1] : kd[s];
          ki[s] = mv ? ki[s - 1] : ki[s];
        }
#pragma unroll
        for (int s = 0; s < KNN; ++s)
          if (s == pos) { kd[s] = d; ki[s] = cand; }
      }
    }
  }
#pragma unroll
  for (int s = 0; s < KNN; ++s) IDX[(size_t)q * KNN + s] = ki[s];
}

// ---------------------------------------------------------------------------
// Edge feature gather: e = [x_i, x_j - x_i], converted to f16, K padded to Kp
// ---------------------------------------------------------------------------
__global__ void gather_edges(const float* __restrict__ X, const int* __restrict__ IDX,
                             _Float16* __restrict__ E, int D, int Kp, int total) {
  int t = blockIdx.x * 256 + threadIdx.x;
  if (t >= total) return;
  int r = t / Kp, c = t - r * Kp;
  int row = r / KNN;                  // global node row
  int b   = row >> 12;
  float v = 0.f;
  if (c < D) {
    v = X[(size_t)row * D + c];
  } else if (c < 2 * D) {
    int j  = IDX[r];
    int cc = c - D;
    v = X[(size_t)((b << 12) + j) * D + cc] - X[(size_t)row * D + cc];
  }
  E[t] = (_Float16)v;
}

// ---------------------------------------------------------------------------
// Weight prep: fp32 [Kin,Nout] -> f16 transposed [Nout,Kp] (zero-padded K)
// ---------------------------------------------------------------------------
__global__ void wt_convert(const float* __restrict__ W, _Float16* __restrict__ Wt,
                           int Kin, int Kp, int Nout, int total) {
  int t = blockIdx.x * 256 + threadIdx.x;
  if (t >= total) return;
  int n = t / Kp, k = t - n * Kp;
  Wt[t] = (k < Kin) ? (_Float16)W[(size_t)k * Nout + n] : (_Float16)0.f;
}

// ---------------------------------------------------------------------------
// GEMM: Y[M,Nout] = ReLU(A[M,Kp](f16) * Wt^T + bias), f32 accumulate via
// v_wmma_f32_16x16x32_f16, plus per-channel sum/sumsq reduction for BN.
// Block = 128 threads (4 waves) -> 128x64 tile; wave -> 32x64 (2 strips).
// Per K-step: stage B tile in LDS, load 2 A frags + 4 B frags into distinct
// registers, then 8 back-to-back WMMAs (each B frag reused twice).
// ---------------------------------------------------------------------------
__global__ __launch_bounds__(128) void gemm_relu_stats(
    const _Float16* __restrict__ A, const _Float16* __restrict__ Wt,
    const float* __restrict__ bias, float* __restrict__ Y,
    float* __restrict__ gsum, float* __restrict__ gsq, int Kp, int Nout) {
  __shared__ __align__(64) _Float16 Bs[64 * 32];
  __shared__ float s_sum[64], s_sq[64];

  const int tid  = threadIdx.x;
  const int lane = tid & 31;
  const int wave = tid >> 5;
  const int m0 = blockIdx.x * 128 + wave * 32;  // wave owns rows m0..m0+31
  const int n0 = blockIdx.y * 64;
  const int lm = lane & 15;       // column / row-in-half selector
  const int hi = lane >> 4;       // lane half

  if (tid < 64) { s_sum[tid] = 0.f; s_sq[tid] = 0.f; }

  v8f acc[2][4];
#pragma unroll
  for (int jt = 0; jt < 4; ++jt) {
    float bv = bias[n0 + jt * 16 + lm];
#pragma unroll
    for (int r = 0; r < 8; ++r) { acc[0][jt][r] = bv; acc[1][jt][r] = bv; }
  }

  // A lane layout (16-bit A 16x32): lanes 0-15 hold K {0..7,16..23} of row lm,
  // lanes 16-31 hold K {8..15,24..31}.
  const _Float16* ap0 = A + (size_t)(m0 + lm) * Kp + hi * 8;
  const _Float16* ap1 = ap0 + (size_t)16 * Kp;

  for (int k0 = 0; k0 < Kp; k0 += 32) {
    __syncthreads();
    {  // stage Wt tile [64 cols][32 K] in LDS: 2048 halves, 16 per thread
      const int e  = tid * 16;
      const int bn = e >> 5;
      const int bk = e & 31;
      const _Float16* src = Wt + (size_t)(n0 + bn) * Kp + k0 + bk;
      *(v8h*)&Bs[e]     = *(const v8h*)src;
      *(v8h*)&Bs[e + 8] = *(const v8h*)(src + 8);
    }
    __syncthreads();

    v8h a0lo = *(const v8h*)(ap0 + k0);
    v8h a0hi = *(const v8h*)(ap0 + k0 + 16);
    v8h a1lo = *(const v8h*)(ap1 + k0);
    v8h a1hi = *(const v8h*)(ap1 + k0 + 16);
    v16h a0 = __builtin_shufflevector(a0lo, a0hi, 0, 1, 2, 3, 4, 5, 6, 7, 8, 9,
                                      10, 11, 12, 13, 14, 15);
    v16h a1 = __builtin_shufflevector(a1lo, a1hi, 0, 1, 2, 3, 4, 5, 6, 7, 8, 9,
                                      10, 11, 12, 13, 14, 15);
    if (k0 + 32 < Kp) {
      __builtin_prefetch(ap0 + k0 + 32, 0, 1);
      __builtin_prefetch(ap1 + k0 + 32, 0, 1);
    }

    // Load all 4 B fragments into distinct registers first so the ds_loads
    // clause together and the 8 WMMAs issue back-to-back.
    v16h bf[4];
#pragma unroll
    for (int jt = 0; jt < 4; ++jt) {
      // B lane layout (16-bit B 32x16): col = lm, K = {8hi..8hi+7, 16+8hi..16+8hi+7}
      const _Float16* bp = &Bs[(jt * 16 + lm) * 32 + hi * 8];
      v8h blo = *(const v8h*)bp;
      v8h bhi = *(const v8h*)(bp + 16);
      bf[jt] = __builtin_shufflevector(blo, bhi, 0, 1, 2, 3, 4, 5, 6, 7, 8, 9,
                                       10, 11, 12, 13, 14, 15);
    }
#pragma unroll
    for (int jt = 0; jt < 4; ++jt) {
      acc[0][jt] = WMMA_F16(a0, bf[jt], acc[0][jt]);
      acc[1][jt] = WMMA_F16(a1, bf[jt], acc[1][jt]);
    }
  }

  // Epilogue: ReLU, store f32, reduce per-channel stats (C/D layout:
  // VGPR r -> row strip + r + 8*hi, col n0 + jt*16 + lm).
#pragma unroll
  for (int jt = 0; jt < 4; ++jt) {
    float s = 0.f, qq = 0.f;
#pragma unroll
    for (int st = 0; st < 2; ++st) {
#pragma unroll
      for (int r = 0; r < 8; ++r) {
        float v = acc[st][jt][r];
        v = v > 0.f ? v : 0.f;
        Y[(size_t)(m0 + st * 16 + r + hi * 8) * Nout + (n0 + jt * 16 + lm)] = v;
        s += v;
        qq += v * v;
      }
    }
    atomicAdd(&s_sum[jt * 16 + lm], s);
    atomicAdd(&s_sq[jt * 16 + lm], qq);
  }
  __syncthreads();
  if (tid < 64) {
    atomicAdd(&gsum[n0 + tid], s_sum[tid]);
    atomicAdd(&gsq[n0 + tid], s_sq[tid]);
  }
}

// ---------------------------------------------------------------------------
// BN: stats -> per-channel affine (biased variance, training mode)
// ---------------------------------------------------------------------------
__global__ void bn_finalize(const float* __restrict__ sum, const float* __restrict__ sq,
                            const float* __restrict__ gamma, const float* __restrict__ beta,
                            float* __restrict__ scale, float* __restrict__ shift,
                            int C, float Mf) {
  int c = blockIdx.x * 256 + threadIdx.x;
  if (c >= C) return;
  float mu  = sum[c] / Mf;
  float var = sq[c] / Mf - mu * mu;
  float rs  = rsqrtf(var + EPS_BN);
  float sc  = gamma[c] * rs;
  scale[c] = sc;
  shift[c] = beta[c] - mu * sc;
}

// BN-apply + f16 requantize for the next WMMA layer
__global__ void normalize_f16(const float* __restrict__ Y, const float* __restrict__ scale,
                              const float* __restrict__ shift, _Float16* __restrict__ Xn,
                              int C, int total) {
  int t = blockIdx.x * 256 + threadIdx.x;
  if (t >= total) return;
  int c = t % C;
  Xn[t] = (_Float16)(scale[c] * Y[t] + shift[c]);
}

// BN-apply + max over k neighbors (exact for negative gamma via min branch);
// writes f32 state for next conv's kNN and the f16 head-concat slice.
__global__ void aggregate_max(const float* __restrict__ Y2, const float* __restrict__ scale,
                              const float* __restrict__ shift, float* __restrict__ Xout,
                              _Float16* __restrict__ H, int slice) {
  int t = blockIdx.x * 256 + threadIdx.x;
  if (t >= BNODE * 64) return;
  int r = t >> 6, c = t & 63;
  const float* p = Y2 + (size_t)r * KNN * 64 + c;
  float mx = -3.4e38f, mn = 3.4e38f;
#pragma unroll
  for (int k = 0; k < KNN; ++k) {
    float v = p[k * 64];
    mx = fmaxf(mx, v);
    mn = fminf(mn, v);
  }
  float sc  = scale[c];
  float val = (sc >= 0.f ? sc * mx : sc * mn) + shift[c];
  Xout[t] = val;
  H[(size_t)r * 192 + slice + c] = (_Float16)val;
}

__global__ void final_out(const float* __restrict__ Y, const float* __restrict__ scale,
                          const float* __restrict__ shift, float* __restrict__ out) {
  int t = blockIdx.x * 256 + threadIdx.x;
  if (t >= BNODE * 128) return;
  int c = t & 127;
  out[t] = scale[c] * Y[t] + shift[c];
}

// ---------------------------------------------------------------------------
// Host orchestration
// ---------------------------------------------------------------------------
extern "C" void kernel_launch(void* const* d_in, const int* in_sizes, int n_in,
                              void* d_out, int out_size, void* d_ws, size_t ws_size,
                              hipStream_t stream) {
  const float* pts = (const float*)d_in[0];
  const float *Wf[10], *bsv[10], *gm[10], *bt[10];
  for (int l = 0; l < 10; ++l) {
    Wf[l]  = (const float*)d_in[1 + 4 * l];
    bsv[l] = (const float*)d_in[2 + 4 * l];
    gm[l]  = (const float*)d_in[3 + 4 * l];
    bt[l]  = (const float*)d_in[4 + 4 * l];
  }
  float* out = (float*)d_out;

  // layer tables: 0-1 conv1, 2-3 conv2, 4-5 conv3, 6-9 head
  const int KiL[10] = {12, 64, 128, 64, 128, 64, 192, 1024, 512, 256};
  const int KpL[10] = {32, 64, 128, 64, 128, 64, 192, 1024, 512, 256};
  const int NoL[10] = {64, 64, 64, 64, 64, 64, 1024, 512, 256, 128};

  // deterministic bump allocator over d_ws
  char* ws = (char*)d_ws;
  size_t off = 0;
  auto alloc = [&](size_t bytes) -> char* {
    char* p = ws + off;
    off = (off + bytes + 255) & ~(size_t)255;
    return p;
  };
  float* X0  = (float*)alloc((size_t)BNODE * 6 * 4);
  float* XA  = (float*)alloc((size_t)BNODE * 64 * 4);
  float* XB  = (float*)alloc((size_t)BNODE * 64 * 4);
  int*   IDX = (int*)alloc((size_t)ME * 4);
  char*  BIG0 = alloc((size_t)ME * 128 * 2);  // edge f16 / conv y2 / head yB
  char*  BIG1 = alloc((size_t)ME * 64 * 4);   // conv y1 / head yA
  char*  BIG2 = alloc((size_t)ME * 64 * 2);   // f16 activations (conv & head)
  _Float16* H16 = (_Float16*)alloc((size_t)BNODE * 192 * 2);
  float* SUM = (float*)alloc(1024 * 4);       // SUM/SQ contiguous: one memset
  float* SQ  = (float*)alloc(1024 * 4);
  float* SCL = (float*)alloc(1024 * 4);
  float* SHF = (float*)alloc(1024 * 4);
  _Float16* WT[10];
  for (int l = 0; l < 10; ++l) WT[l] = (_Float16*)alloc((size_t)NoL[l] * KpL[l] * 2);

  // weight conversion (cheap; redone each call for determinism)
  for (int l = 0; l < 10; ++l) {
    int total = NoL[l] * KpL[l];
    wt_convert<<<(total + 255) / 256, 256, 0, stream>>>(Wf[l], WT[l], KiL[l],
                                                        KpL[l], NoL[l], total);
  }

  build_x0<<<(BNODE * 6 + 255) / 256, 256, 0, stream>>>(pts, X0);

  auto run_conv = [&](const float* xin, int D, int L1, int L2, float* xout, int slice) {
    if (D == 6)
      knn_kernel<6><<<BNODE / 128, 128, 0, stream>>>(xin, IDX);
    else
      knn_kernel<64><<<BNODE / 128, 128, 0, stream>>>(xin, IDX);
    int Kp1  = KpL[L1];
    int totE = ME * Kp1;
    gather_edges<<<(totE + 255) / 256, 256, 0, stream>>>(xin, IDX, (_Float16*)BIG0,
                                                         D, Kp1, totE);
    hipMemsetAsync(SUM, 0, 2 * 1024 * 4, stream);
    gemm_relu_stats<<<dim3(ME / 128, 1), 128, 0, stream>>>(
        (const _Float16*)BIG0, WT[L1], bsv[L1], (float*)BIG1, SUM, SQ, Kp1, 64);
    bn_finalize<<<1, 256, 0, stream>>>(SUM, SQ, gm[L1], bt[L1], SCL, SHF, 64, (float)ME);
    int totN = ME * 64;
    normalize_f16<<<(totN + 255) / 256, 256, 0, stream>>>((const float*)BIG1, SCL,
                                                          SHF, (_Float16*)BIG2, 64, totN);
    hipMemsetAsync(SUM, 0, 2 * 1024 * 4, stream);
    gemm_relu_stats<<<dim3(ME / 128, 1), 128, 0, stream>>>(
        (const _Float16*)BIG2, WT[L2], bsv[L2], (float*)BIG0, SUM, SQ, 64, 64);
    bn_finalize<<<1, 256, 0, stream>>>(SUM, SQ, gm[L2], bt[L2], SCL, SHF, 64, (float)ME);
    aggregate_max<<<(BNODE * 64 + 255) / 256, 256, 0, stream>>>(
        (const float*)BIG0, SCL, SHF, xout, H16, slice);
  };

  run_conv(X0, 6, 0, 1, XA, 0);
  run_conv(XA, 64, 2, 3, XB, 64);
  run_conv(XB, 64, 4, 5, XA, 128);

  // head: ping-pong BIG1 (yA) / BIG0 (yB), f16 activations in BIG2
  float* YA = (float*)BIG1;
  float* YB = (float*)BIG0;
  _Float16* XN = (_Float16*)BIG2;

  auto run_head = [&](const _Float16* Ain, int L, float* Yb, bool last) {
    hipMemsetAsync(SUM, 0, 2 * 1024 * 4, stream);
    gemm_relu_stats<<<dim3(BNODE / 128, NoL[L] / 64), 128, 0, stream>>>(
        Ain, WT[L], bsv[L], Yb, SUM, SQ, KpL[L], NoL[L]);
    bn_finalize<<<(NoL[L] + 255) / 256, 256, 0, stream>>>(SUM, SQ, gm[L], bt[L],
                                                          SCL, SHF, NoL[L], (float)BNODE);
    if (!last) {
      int tot = BNODE * NoL[L];
      normalize_f16<<<(tot + 255) / 256, 256, 0, stream>>>(Yb, SCL, SHF, XN, NoL[L], tot);
    } else {
      final_out<<<(BNODE * 128 + 255) / 256, 256, 0, stream>>>(Yb, SCL, SHF, out);
    }
  };

  run_head(H16, 6, YA, false);
  run_head(XN, 7, YB, false);
  run_head(XN, 8, YA, false);
  run_head(XN, 9, YB, true);
}
